// DiffusionTetraChirality_82841329205440
// MI455X (gfx1250) — matrix operand order repaired and checked by reading, not compiled, not run on hardware
//
#include <hip/hip_runtime.h>

// DiffusionTetraChirality on MI455X (gfx1250, wave32, WMMA).
// One wave per perm row; MLP (16 x 259->64->64->64->2) via v_wmma_f32_16x16x32_f16.
// Weights pre-swizzled into LDS in WMMA B-fragment order so each operand is
// two ds_load_b128 (32 contiguous bytes per lane) instead of 16 ds_load_u16.
// Activations staged row-major in a per-wave LDS tile (A-frags = 2x b128).
// Scatter-add via global_atomic_add_f32 into a copy of `answer`.

#define TT    8192          // T
#define SS    16            // S
#define DD    64            // D
#define DIN   259
#define NPERM (3*TT)        // 24576 perm rows
#define LEAKY 1.0e-3f

#define NF1   36            // layer-1 fragments: 9 K-steps x 4 N-tiles
#define NF2   8             // layer-2/3 fragments: 2 K-steps x 4 N-tiles
#define FRAG  512           // 32 lanes x 16 halfwords per fragment

typedef __attribute__((ext_vector_type(16))) _Float16 v16h;
typedef __attribute__((ext_vector_type(8)))  _Float16 v8h;
typedef __attribute__((ext_vector_type(8)))  float    v8f;
typedef __attribute__((ext_vector_type(4)))  float    v4f;

__global__ void dtc_copy(const float* __restrict__ src, float* __restrict__ dst, int n4) {
  int i = blockIdx.x * blockDim.x + threadIdx.x;
  if (i < n4) ((v4f*)dst)[i] = ((const v4f*)src)[i];
}

__global__ __launch_bounds__(256)
void dtc_main(const float* __restrict__ coords,
              const int*   __restrict__ tetras,
              const float* __restrict__ encoded,
              const float* __restrict__ tbuf,
              const float* __restrict__ W1, const float* __restrict__ b1,
              const float* __restrict__ W2, const float* __restrict__ b2,
              const float* __restrict__ W3, const float* __restrict__ b3,
              const float* __restrict__ W4, const float* __restrict__ b4,
              float* __restrict__ out)
{
  // Fragment-order packed weights: w?p[f*512 + lane*16 + e]
  __shared__ _Float16 w1p[NF1*FRAG];  // 36,864 B
  __shared__ _Float16 w2p[NF2*FRAG];  //  8,192 B
  __shared__ _Float16 w3p[NF2*FRAG];  //  8,192 B
  __shared__ _Float16 hsh[8*SS*DD];   // 16,384 B (per-wave 16x64 row-major tile)

  // ---- cooperative prepack: f32 global -> f16 LDS in B-fragment order ----
  // fragment f = kk*4 + nt; lane element e covers K = kk*32 + (e>=8?16:0) + ((lane>>4)<<3) + (e&7),
  // column = nt*16 + (lane&15).  (K rows >= DIN are zero-padded.)
  for (int j = threadIdx.x; j < NF1*FRAG; j += 256) {
    const int e = j & 15, ln = (j >> 4) & 31, f = j >> 9;
    const int k   = (f >> 2)*32 + ((e >> 3) << 4) + ((ln >> 4) << 3) + (e & 7);
    const int col = (f & 3)*16 + (ln & 15);
    w1p[j] = (k < DIN) ? (_Float16)W1[k*DD + col] : (_Float16)0.f;
  }
  for (int j = threadIdx.x; j < NF2*FRAG; j += 256) {
    const int e = j & 15, ln = (j >> 4) & 31, f = j >> 9;
    const int k   = (f >> 2)*32 + ((e >> 3) << 4) + ((ln >> 4) << 3) + (e & 7);
    const int col = (f & 3)*16 + (ln & 15);
    w2p[j] = (_Float16)W2[k*DD + col];
    w3p[j] = (_Float16)W3[k*DD + col];
  }
  __syncthreads();

  const int wave = threadIdx.x >> 5;
  const int lane = threadIdx.x & 31;
  const int m    = lane & 15;          // A-frag row (= sample s) / C column
  const int hi8  = (lane >> 4) << 3;   // 0|8 : A-frag K offset, C-layout row offset
  _Float16* hb = hsh + wave * (SS*DD);

  const int p      = blockIdx.x * 8 + wave;   // perm row id [0, 24576)
  const int permId = p >> 13;                 // p / TT  (TT = 8192)
  const int tet    = p & (TT - 1);

  const int t0 = tetras[tet*5+0];
  const int t1 = tetras[tet*5+1];
  const int t2 = tetras[tet*5+2];
  const int t3 = tetras[tet*5+3];
  const float sgn = (float)tetras[tet*5+4];

  int ia0 = t0, ia1, ia2, ia3;
  if (permId == 0)      { ia1 = t1; ia2 = t2; ia3 = t3; }
  else if (permId == 1) { ia1 = t3; ia2 = t1; ia3 = t2; }
  else                  { ia1 = t2; ia2 = t3; ia3 = t1; }

  // ---- geometry for s = m (lanes 16..31 recompute the same values) ----
  const float* c0 = coords + ((long)ia0*SS + m)*3;
  const float* c1 = coords + ((long)ia1*SS + m)*3;
  const float* c2 = coords + ((long)ia2*SS + m)*3;
  const float* c3 = coords + ((long)ia3*SS + m)*3;
  float p0x = c0[0], p0y = c0[1], p0z = c0[2];
  float v0x = c1[0]-p0x, v0y = c1[1]-p0y, v0z = c1[2]-p0z;
  float v1x = c2[0]-p0x, v1y = c2[1]-p0y, v1z = c2[2]-p0z;
  float v2x = c3[0]-p0x, v2y = c3[1]-p0y, v2z = c3[2]-p0z;
  float crx = sgn*(v1y*v2z - v1z*v2y);
  float cry = sgn*(v1z*v2x - v1x*v2z);
  float crz = sgn*(v1x*v2y - v1y*v2x);
  float rn = 1.f / sqrtf(crx*crx + cry*cry + crz*crz);
  crx *= rn; cry *= rn; crz *= rn;
  float smx = v1x+v2x, smy = v1y+v2y, smz = v1z+v2z;
  float rs = 1.f / sqrtf(smx*smx + smy*smy + smz*smz);
  smx *= rs; smy *= rs; smz *= rs;
  const float outv   =  crx*v0x + cry*v0y + crz*v0z;
  const float alongv = -(smx*v0x + smy*v0y + smz*v0z);
  const float tval   = tbuf[m];

  const float* e0 = encoded + (long)ia0*DD;
  const float* e1 = encoded + (long)ia1*DD;
  const float* e2 = encoded + (long)ia2*DD;
  const float* e3 = encoded + (long)ia3*DD;

  // ---- layer 1: feats(16x288) @ W1(288x64), f16 WMMA, f32 accumulate ----
  v8f acc[4] = {};
  #pragma unroll
  for (int kk = 0; kk < 9; ++kk) {
    v16h A{};
    #pragma unroll
    for (int e = 0; e < 16; ++e) {
      const int v = e >> 1, hh = e & 1;
      const int kl = ((v >= 4) ? 16 : 0) + hi8 + ((v & 3) << 1) + hh;  // 0..31
      float val;
      if (kk < 8) {
        // columns 0..255: enc[idx[k>>6]][k&63]; (kk*32+kl)>>6 == kk>>1 (const after unroll)
        const float* er = (kk < 2) ? e0 : (kk < 4) ? e1 : (kk < 6) ? e2 : e3;
        val = er[((kk & 1) << 5) + kl];     // two contiguous 8-float runs -> global_load_b128
      } else {
        // columns 256..287: [t, out/4, along/4, 0...]
        val = (kl == 0) ? tval : (kl == 1) ? outv*0.25f : (kl == 2) ? alongv*0.25f : 0.f;
      }
      A[e] = (_Float16)val;
    }
    #pragma unroll
    for (int nt = 0; nt < 4; ++nt) {
      // one 32-byte contiguous chunk per lane -> 2x ds_load_b128
      const v16h B = *(const v16h*)(w1p + ((kk*4 + nt)*32 + lane)*16);
      acc[nt] = __builtin_amdgcn_wmma_f32_16x16x32_f16(
          false, A, false, B, (short)0, acc[nt], false, false);
    }
  }
  // bias + leaky relu -> per-wave LDS tile (row-major 16x64, f16)
  #pragma unroll
  for (int nt = 0; nt < 4; ++nt) {
    #pragma unroll
    for (int r = 0; r < 8; ++r) {
      const int row = r + hi8;
      const int col = nt*16 + m;
      float x = acc[nt][r] + b1[col];
      x = (x >= 0.f) ? x : LEAKY*x;
      hb[row*DD + col] = (_Float16)x;
    }
  }
  __syncthreads();

  // ---- layers 2 and 3: h(16x64) @ W(64x64) ----
  auto run_layer = [&](const _Float16* __restrict__ wp, const float* __restrict__ bias) {
    v8f a2[4] = {};
    #pragma unroll
    for (int kk = 0; kk < 2; ++kk) {
      // A elements 0..7 and 8..15 are two contiguous 16B runs in the row-major tile
      const v8h alo = *(const v8h*)(hb + m*DD + kk*32 + hi8);
      const v8h ahi = *(const v8h*)(hb + m*DD + kk*32 + 16 + hi8);
      const v16h A = __builtin_shufflevector(alo, ahi,
          0,1,2,3,4,5,6,7,8,9,10,11,12,13,14,15);
      #pragma unroll
      for (int nt = 0; nt < 4; ++nt) {
        const v16h B = *(const v16h*)(wp + ((kk*4 + nt)*32 + lane)*16);
        a2[nt] = __builtin_amdgcn_wmma_f32_16x16x32_f16(
            false, A, false, B, (short)0, a2[nt], false, false);
      }
    }
    __syncthreads();   // all reads of hb done before overwrite
    #pragma unroll
    for (int nt = 0; nt < 4; ++nt) {
      #pragma unroll
      for (int r = 0; r < 8; ++r) {
        const int row = r + hi8;
        const int col = nt*16 + m;
        float x = a2[nt][r] + bias[col];
        x = (x >= 0.f) ? x : LEAKY*x;
        hb[row*DD + col] = (_Float16)x;
      }
    }
    __syncthreads();   // writes visible to all lanes
  };
  run_layer(w2p, b2);
  run_layer(w3p, b3);

  // ---- layer 4 (64 -> 2) + scatter-add (lanes 0..15, one sample each) ----
  if (lane < 16) {
    float d0 = b4[0], d1 = b4[1];
    const v8h* hrow = (const v8h*)(hb + m*DD);   // contiguous row -> ds_load_b128 x8
    #pragma unroll
    for (int c8 = 0; c8 < 8; ++c8) {
      const v8h hv8 = hrow[c8];
      #pragma unroll
      for (int j = 0; j < 8; ++j) {
        const float hv = (float)hv8[j];
        const int c = c8*8 + j;
        d0 = fmaf(hv, W4[c*2+0], d0);
        d1 = fmaf(hv, W4[c*2+1], d1);
      }
    }
    const float s0 = -0.25f * d0;
    const float s1 =  0.25f * d1;
    const long o0 = ((long)ia0*SS + m)*3;
    const long o1 = ((long)ia1*SS + m)*3;
    atomicAdd(&out[o0+0], s0*crx);
    atomicAdd(&out[o0+1], s0*cry);
    atomicAdd(&out[o0+2], s0*crz);
    atomicAdd(&out[o1+0], s1*crx);
    atomicAdd(&out[o1+1], s1*cry);
    atomicAdd(&out[o1+2], s1*crz);
  }
}

extern "C" void kernel_launch(void* const* d_in, const int* in_sizes, int n_in,
                              void* d_out, int out_size, void* d_ws, size_t ws_size,
                              hipStream_t stream) {
  (void)in_sizes; (void)n_in; (void)d_ws; (void)ws_size;
  const float* coords  = (const float*)d_in[0];
  const int*   tetras  = (const int*)  d_in[1];
  const float* encoded = (const float*)d_in[2];
  const float* tbuf    = (const float*)d_in[3];
  const float* answer  = (const float*)d_in[4];
  const float* W1 = (const float*)d_in[5];
  const float* b1 = (const float*)d_in[6];
  const float* W2 = (const float*)d_in[7];
  const float* b2 = (const float*)d_in[8];
  const float* W3 = (const float*)d_in[9];
  const float* b3 = (const float*)d_in[10];
  const float* W4 = (const float*)d_in[11];
  const float* b4 = (const float*)d_in[12];
  float* out = (float*)d_out;

  // initialize output with `answer` (N*S*3 = 2.4M floats, divisible by 4)
  const int n4 = out_size >> 2;
  dtc_copy<<<(n4 + 255)/256, 256, 0, stream>>>(answer, out, n4);
  // 24576 perms, 8 waves (one perm each) per 256-thread block
  dtc_main<<<NPERM/8, 256, 0, stream>>>(coords, tetras, encoded, tbuf,
                                        W1, b1, W2, b2, W3, b3, W4, b4, out);
}
